// L2Dist_42013370089990
// MI455X (gfx1250) — matrix operand order repaired and checked by reading, not compiled, not run on hardware
//
#include <hip/hip_runtime.h>
#include <hip/hip_bf16.h>

typedef __attribute__((ext_vector_type(16))) __bf16 v16bf;
typedef __attribute__((ext_vector_type(8)))  float  v8f;

#define D       128     // feature dim (reference: d=128)
#define NTILE   256     // pred columns per workgroup
#define THREADS 256     // 8 waves of 32
#define NPOS_MAX 2048   // target rows (reference: 2048)

// ---------- bf16 helpers (manual RNE split: f = hi + lo in bf16) ----------
__device__ __forceinline__ unsigned short f2bf_rne(float f) {
  unsigned int u = __builtin_bit_cast(unsigned int, f);
  u += 0x7FFFu + ((u >> 16) & 1u);
  return (unsigned short)(u >> 16);
}
__device__ __forceinline__ float bf2f(unsigned short h) {
  unsigned int u = ((unsigned int)h) << 16;
  return __builtin_bit_cast(float, u);
}

struct U8  { unsigned int u[8]; };   // 32B -> v16bf
struct Q2  { uint4 a, b; };          // 32B -> v16bf

// ---------- prep 1: segment counts ----------
__global__ void seg_count_kernel(const int* __restrict__ ids,
                                 const int* __restrict__ nsegp,
                                 int* __restrict__ counts, int n_pos) {
  __shared__ int c[64];
  int nseg = nsegp[0]; if (nseg > 64) nseg = 64; if (nseg < 1) nseg = 1;
  const int tid = threadIdx.x;
  if (tid < 64) c[tid] = 0;
  __syncthreads();
  for (int i = tid; i < n_pos; i += blockDim.x) {
    int id = ids[i];
    if (id >= 0 && id < nseg) atomicAdd(&c[id], 1);
  }
  __syncthreads();
  if (tid < nseg) counts[tid] = c[tid];
}

// ---------- prep 2: pack target rows into WMMA A-fragment layout ----------
// Layout: slot(c,kk,g,m) = ((c*4+kk)*4+g)*16+m holds 8 consecutive bf16
// (K = kk*32 + g*8 .. +7) for target row p = c*16+m. One 16B load per
// half-fragment in the main loop.
__global__ void prep_pack_kernel(const float* __restrict__ target,
                                 const int* __restrict__ ids,
                                 const int* __restrict__ counts,
                                 const int* __restrict__ nsegp,
                                 unsigned short* __restrict__ thi,
                                 unsigned short* __restrict__ tlo,
                                 float* __restrict__ t2,
                                 float* __restrict__ wseg) {
  const int p = blockIdx.x;
  const int k = threadIdx.x;            // 0..127
  float f = target[(size_t)p * D + k];
  unsigned short h = f2bf_rne(f);
  unsigned short l = f2bf_rne(f - bf2f(h));
  const int c = p >> 4, m = p & 15;
  const int kk = k >> 5, r = k & 31, g = r >> 3, e = r & 7;
  const int addr = (((c * 4 + kk) * 4 + g) * 16 + m) * 8 + e;
  thi[addr] = h;
  tlo[addr] = l;

  __shared__ float red[D];
  red[k] = f * f;
  __syncthreads();
  #pragma unroll
  for (int s = D / 2; s > 0; s >>= 1) {
    if (k < s) red[k] += red[k + s];
    __syncthreads();
  }
  if (k == 0) {
    t2[p] = red[0];
    int nseg = nsegp[0]; if (nseg < 1) nseg = 1;
    int id = ids[p];
    int cnt = (id >= 0 && id < nseg) ? counts[id] : 1;
    if (cnt < 1) cnt = 1;
    wseg[p] = 1.0f / ((float)nseg * (float)cnt);
  }
}

// ---------- main: fused GEMM (bf16x2 split) + sqrt + weighted reduce ----------
__global__ void __launch_bounds__(THREADS)
l2seg_main_kernel(const float* __restrict__ pred,
                  const unsigned short* __restrict__ thi,
                  const unsigned short* __restrict__ tlo,
                  const float* __restrict__ t2g,
                  const float* __restrict__ wg,
                  float* __restrict__ out,
                  int n_nodes, int n_pos) {
  __shared__ float s_t2[NPOS_MAX];
  __shared__ float s_w[NPOS_MAX];
  const int tid = threadIdx.x;
  const int wave = tid >> 5;
  const int lane = tid & 31;
  const int lanecol = lane & 15;
  const int half = lane >> 4;           // 0: k 0-7,16-23; 1: k 8-15,24-31 of each 32-k step

  for (int i = tid; i < n_pos; i += THREADS) { s_t2[i] = t2g[i]; s_w[i] = wg[i]; }
  __syncthreads();

  const int n0 = blockIdx.x * NTILE + wave * 32;

  // Build register-resident B fragments (pred columns), bf16 hi/lo split,
  // and per-column squared norms. Pred is read from HBM exactly once.
  v16bf Bh[2][4], Bl[2][4];
  float p2c[2];
  #pragma unroll
  for (int t = 0; t < 2; ++t) {
    int colL = n0 + t * 16 + lanecol;
    int col = colL < n_nodes ? colL : (n_nodes - 1);   // clamp; store masked later
    const float* rowp = pred + (size_t)col * D + half * 8;
    float s = 0.f;
    #pragma unroll
    for (int kk = 0; kk < 4; ++kk) {
      U8 bh, bl;
      #pragma unroll
      for (int j = 0; j < 8; ++j) {
        const float2 f = *(const float2*)(rowp + kk * 32 + 2 * (j & 3) + (j >> 2) * 16);
        s = fmaf(f.x, f.x, s);
        s = fmaf(f.y, f.y, s);
        unsigned short h0 = f2bf_rne(f.x), h1 = f2bf_rne(f.y);
        unsigned short l0 = f2bf_rne(f.x - bf2f(h0));
        unsigned short l1 = f2bf_rne(f.y - bf2f(h1));
        bh.u[j] = (unsigned int)h0 | ((unsigned int)h1 << 16);
        bl.u[j] = (unsigned int)l0 | ((unsigned int)l1 << 16);
      }
      Bh[t][kk] = __builtin_bit_cast(v16bf, bh);
      Bl[t][kk] = __builtin_bit_cast(v16bf, bl);
    }
    s += __shfl_xor(s, 16, 32);
    p2c[t] = s;
  }

  const uint4* __restrict__ thiq = (const uint4*)thi;
  const uint4* __restrict__ tloq = (const uint4*)tlo;

  float accum0 = 0.f, accum1 = 0.f;
  const int nchunks = n_pos >> 4;       // 16 target rows per chunk
  for (int c = 0; c < nchunks; ++c) {
    v8f acc0 = {0.f, 0.f, 0.f, 0.f, 0.f, 0.f, 0.f, 0.f};
    v8f acc1 = {0.f, 0.f, 0.f, 0.f, 0.f, 0.f, 0.f, 0.f};
    #pragma unroll
    for (int kk = 0; kk < 4; ++kk) {
      const int sbase = ((c * 4 + kk) * 4) * 16 + lanecol;
      Q2 qh, ql;
      qh.a = thiq[sbase + half * 16];
      qh.b = thiq[sbase + (2 + half) * 16];
      ql.a = tloq[sbase + half * 16];
      ql.b = tloq[sbase + (2 + half) * 16];
      v16bf Ah = __builtin_bit_cast(v16bf, qh);
      v16bf Al = __builtin_bit_cast(v16bf, ql);
      // cross ~= hi*hi + hi*lo + lo*hi  (f32 accumulate)
      acc0 = __builtin_amdgcn_wmma_f32_16x16x32_bf16(false, Ah, false, Bh[0][kk], (short)0, acc0, false, false);
      acc0 = __builtin_amdgcn_wmma_f32_16x16x32_bf16(false, Ah, false, Bl[0][kk], (short)0, acc0, false, false);
      acc0 = __builtin_amdgcn_wmma_f32_16x16x32_bf16(false, Al, false, Bh[0][kk], (short)0, acc0, false, false);
      acc1 = __builtin_amdgcn_wmma_f32_16x16x32_bf16(false, Ah, false, Bh[1][kk], (short)0, acc1, false, false);
      acc1 = __builtin_amdgcn_wmma_f32_16x16x32_bf16(false, Ah, false, Bl[1][kk], (short)0, acc1, false, false);
      acc1 = __builtin_amdgcn_wmma_f32_16x16x32_bf16(false, Al, false, Bh[1][kk], (short)0, acc1, false, false);
    }
    // Fused epilogue on the accumulator fragment:
    // row M = half*8 + i  ->  p = c*16 + M   (matches D-matrix VGPR layout)
    const int pb = c * 16 + half * 8;
    const float4 wa = *(const float4*)&s_w[pb];
    const float4 wb = *(const float4*)&s_w[pb + 4];
    const float4 ta = *(const float4*)&s_t2[pb];
    const float4 tb = *(const float4*)&s_t2[pb + 4];
    const float w8[8] = {wa.x, wa.y, wa.z, wa.w, wb.x, wb.y, wb.z, wb.w};
    const float t8[8] = {ta.x, ta.y, ta.z, ta.w, tb.x, tb.y, tb.z, tb.w};
    #pragma unroll
    for (int i = 0; i < 8; ++i) {
      float sq0 = fmaf(-2.f, acc0[i], t8[i] + p2c[0]);
      sq0 = fmaxf(sq0, 0.f);
      float r0 = __builtin_amdgcn_sqrtf(sq0 + 1e-8f);
      accum0 = fmaf(w8[i], r0, accum0);
      float sq1 = fmaf(-2.f, acc1[i], t8[i] + p2c[1]);
      sq1 = fmaxf(sq1, 0.f);
      float r1 = __builtin_amdgcn_sqrtf(sq1 + 1e-8f);
      accum1 = fmaf(w8[i], r1, accum1);
    }
  }

  // combine row halves (M 0-7 and 8-15) (column n is held by lane c and lane c+16)
  accum0 += __shfl_xor(accum0, 16, 32);
  accum1 += __shfl_xor(accum1, 16, 32);
  if (half == 0) {
    const int na = n0 + lanecol;
    const int nb = n0 + 16 + lanecol;
    if (na < n_nodes) out[na] = accum0;
    if (nb < n_nodes) out[nb] = accum1;
  }
}

extern "C" void kernel_launch(void* const* d_in, const int* in_sizes, int n_in,
                              void* d_out, int out_size, void* d_ws, size_t ws_size,
                              hipStream_t stream) {
  const float* pred   = (const float*)d_in[0];
  const float* target = (const float*)d_in[1];
  const int*   ids    = (const int*)d_in[2];
  const int*   nsegp  = (const int*)d_in[3];
  float* out = (float*)d_out;

  const int n_nodes = in_sizes[0] / D;
  const int n_pos   = in_sizes[2];

  // workspace layout
  unsigned short* thi = (unsigned short*)d_ws;                    // n_pos*D bf16
  unsigned short* tlo = thi + (size_t)n_pos * D;                  // n_pos*D bf16
  float* t2   = (float*)(tlo + (size_t)n_pos * D);                // n_pos f32
  float* wseg = t2 + n_pos;                                       // n_pos f32
  int*   counts = (int*)(wseg + n_pos);                           // <=64 ints

  seg_count_kernel<<<1, 256, 0, stream>>>(ids, nsegp, counts, n_pos);
  prep_pack_kernel<<<n_pos, D, 0, stream>>>(target, ids, counts, nsegp,
                                            thi, tlo, t2, wseg);
  const int nblocks = (n_nodes + NTILE - 1) / NTILE;
  l2seg_main_kernel<<<nblocks, THREADS, 0, stream>>>(pred, thi, tlo, t2, wseg,
                                                     out, n_nodes, n_pos);
}
